// RegionProposalNetwork_84825604096031
// MI455X (gfx1250) — matrix operand order repaired and checked by reading, not compiled, not run on hardware
//
#include <hip/hip_runtime.h>
#include <hip/hip_bf16.h>

// ---------------------------------------------------------------------------
// RPN: dwconv -> pointwise GEMM (WMMA bf16, async-LDS A-tile, transposed B)
//      -> heads GEMM -> box decode -> bitonic sort -> IoU bitmask -> NMS -> out
// ---------------------------------------------------------------------------

#define C_CH   256
#define HW     1024
#define A_NUM  9
#define NBOX   9216       // A * HW
#define NSORT  16384      // next pow2 >= NBOX
#define MASKW  288        // u32 words per mask row (9216/32)
#define EPS_BN 1e-5f
#define IOU_THR 0.9f

typedef __attribute__((ext_vector_type(16))) __bf16 v16bf;
typedef __attribute__((ext_vector_type(8)))  float  v8f;

union Frag16 {
    v16bf v;
    uint4 u[2];
    unsigned short s[16];
};

__device__ __forceinline__ unsigned short f32_to_bf16(float f) {
    unsigned int u = __float_as_uint(f);
    u += 0x7FFFu + ((u >> 16) & 1u);   // round-to-nearest-even
    return (unsigned short)(u >> 16);
}

// ---------------------------------------------------------------------------
// Stage one 16x256 bf16 weight tile (8 KB) global -> LDS using the CDNA5
// async-to-LDS DMA path (tracked by ASYNCcnt), then barrier.
// 256 threads x 2 chunks x 16 B.
// ---------------------------------------------------------------------------
__device__ __forceinline__ void async_stage_tileA(
    const unsigned short* __restrict__ gA, unsigned short* sA, int tid)
{
#pragma unroll
    for (int it = 0; it < 2; ++it) {
        int elem = (tid + it * 256) * 8;              // 8 bf16 = 16 B per lane
        unsigned lds_off = (unsigned)(size_t)(sA + elem);   // low 32b = LDS addr
        const unsigned short* src = gA + elem;
        asm volatile("global_load_async_to_lds_b128 %0, %1, off"
                     :: "v"(lds_off), "v"(src) : "memory");
    }
    asm volatile("s_wait_asynccnt 0" ::: "memory");
    __syncthreads();
}

// ---------------------------------------------------------------------------
// One 16(M)x16(N) tile, K=256 bf16 GEMM via v_wmma_f32_16x16x32_bf16.
// A: LDS tile [16][256] (row-major, K contiguous)  -> two ds_load_b128 / step
// B: transposed activations BT[N][256] (K contiguous) -> two b128 / step
// A-frag: lanes 0-15 M=lane, K in {k0+8h..+7, k0+16+8h..+23}
// B-frag: lane holds fixed N, K in {k0+8h..+7, k0+16+8h..+23}
// ---------------------------------------------------------------------------
__device__ __forceinline__ v8f gemm_tile_bf16(
    const unsigned short* __restrict__ sA,
    const unsigned short* __restrict__ BT, int n, int lane)
{
    const int half = lane >> 4;     // 0 or 1
    const int l15  = lane & 15;
    v8f acc = {};
    const unsigned short* ar = sA + l15 * 256 + half * 8;
    const unsigned short* br = BT + n * 256 + half * 8;
    __builtin_prefetch(br, 0, 3);                 // global_prefetch_b8
#pragma unroll
    for (int k0 = 0; k0 < 256; k0 += 32) {
        Frag16 a, b;
        a.u[0] = *(const uint4*)(ar + k0);        // K = k0+8h   .. +7
        a.u[1] = *(const uint4*)(ar + k0 + 16);   // K = k0+8h+16.. +23
        b.u[0] = *(const uint4*)(br + k0);        // K = k0+8h   .. +7
        b.u[1] = *(const uint4*)(br + k0 + 16);   // K = k0+8h+16.. +23
        acc = __builtin_amdgcn_wmma_f32_16x16x32_bf16(
            false, a.v, false, b.v, (short)0, acc, false, false);
    }
    return acc;
}

// ---------------------------------------------------------------------------
// Prep: convert pw weights to bf16; pack anc(36)+pad+obj(9)+pad into a padded
// 64x256 bf16 head-weight matrix; zero the keep flags.
// ---------------------------------------------------------------------------
__global__ void k_prep(const float* __restrict__ pww,
                       const float* __restrict__ ancw,
                       const float* __restrict__ objw,
                       unsigned short* __restrict__ wpw,
                       unsigned short* __restrict__ whead,
                       unsigned int* __restrict__ keep)
{
    int t = blockIdx.x * blockDim.x + threadIdx.x;     // 0..65535
    if (t < C_CH * C_CH) wpw[t] = f32_to_bf16(pww[t]);
    if (t < 64 * C_CH) {
        int row = t >> 8, col = t & 255;
        float v = 0.0f;
        if (row < 36)                   v = ancw[row * C_CH + col];
        else if (row >= 48 && row < 57) v = objw[(row - 48) * C_CH + col];
        whead[t] = f32_to_bf16(v);
    }
    if (t < NBOX) keep[t] = 0u;
}

// ---------------------------------------------------------------------------
// Depthwise 3x3 conv, zero pad.  Output TRANSPOSED: h1T[hw][c] bf16
// (thread = channel-major so the h1T store is coalesced).
// ---------------------------------------------------------------------------
__global__ void k_dwconv(const float* __restrict__ x,
                         const float* __restrict__ w,
                         unsigned short* __restrict__ h1T)
{
    int t = blockIdx.x * blockDim.x + threadIdx.x;   // 0..262143
    int c = t & 255, hw = t >> 8;
    int y = hw >> 5, xc = hw & 31;
    const float* xp = x + c * HW;
    const float* wp = w + c * 9;
    float acc = 0.0f;
#pragma unroll
    for (int dy = -1; dy <= 1; ++dy) {
        int yy = y + dy;
        if ((unsigned)yy < 32u) {
#pragma unroll
            for (int dx = -1; dx <= 1; ++dx) {
                int xx = xc + dx;
                if ((unsigned)xx < 32u)
                    acc = fmaf(xp[yy * 32 + xx], wp[(dy + 1) * 3 + (dx + 1)], acc);
            }
        }
    }
    h1T[hw * C_CH + c] = f32_to_bf16(acc);
}

// ---------------------------------------------------------------------------
// GEMM1: h2T = leaky_relu(bn1(pw_w @ h1))^T.  M=256 N=1024 K=256.
// Block = 8 waves sharing one async-staged 16-row weight tile.
// grid 128: mtile = blk>>3 (0..15), ntile = (blk&7)*8 + wave (0..63).
// ---------------------------------------------------------------------------
__global__ void k_gemm1(const unsigned short* __restrict__ Wpw,
                        const unsigned short* __restrict__ h1T,
                        const float* __restrict__ g, const float* __restrict__ b,
                        const float* __restrict__ mu, const float* __restrict__ var,
                        unsigned short* __restrict__ h2T)
{
    __shared__ unsigned short sA[16 * 256];
    int tid = threadIdx.x;
    int mtile = blockIdx.x >> 3;
    async_stage_tileA(Wpw + mtile * 16 * C_CH, sA, tid);

    int wv = tid >> 5, lane = tid & 31;
    int ntile = (blockIdx.x & 7) * 8 + wv;
    int half = lane >> 4, l15 = lane & 15;
    int n = ntile * 16 + l15;
    v8f acc = gemm_tile_bf16(sA, h1T, n, lane);

    union { uint4 u; unsigned short s[8]; } pk;
#pragma unroll
    for (int r = 0; r < 8; ++r) {
        int o = mtile * 16 + 8 * half + r;
        float v = acc[r];
        float inv = rsqrtf(var[o] + EPS_BN);
        v = (v - mu[o]) * (inv * g[o]) + b[o];
        v = v > 0.0f ? v : 0.01f * v;
        pk.s[r] = f32_to_bf16(v);
    }
    // 8 contiguous bf16 -> one 16B store into transposed activations
    *(uint4*)(h2T + n * C_CH + mtile * 16 + 8 * half) = pk.u;
}

// ---------------------------------------------------------------------------
// Heads: one padded 64x256 GEMM covers anc (rows 0-35) and obj (rows 48-56).
// grid 32: mtile = blk>>3 (0..3), ntile = (blk&7)*8 + wave.
// Epilogue: BN + clip + box decode  /  BN + sigmoid -> scores.
// ---------------------------------------------------------------------------
__global__ void k_heads(const unsigned short* __restrict__ Wh,
                        const unsigned short* __restrict__ h2T,
                        const float* __restrict__ anchors,
                        const float* __restrict__ ag, const float* __restrict__ ab,
                        const float* __restrict__ am, const float* __restrict__ av,
                        const float* __restrict__ og, const float* __restrict__ ob,
                        const float* __restrict__ om, const float* __restrict__ ov,
                        float* __restrict__ boxes, float* __restrict__ scores)
{
    __shared__ unsigned short sA[16 * 256];
    int tid = threadIdx.x;
    int mtile = blockIdx.x >> 3;
    async_stage_tileA(Wh + mtile * 16 * C_CH, sA, tid);

    int wv = tid >> 5, lane = tid & 31;
    int ntile = (blockIdx.x & 7) * 8 + wv;
    int half = lane >> 4, l15 = lane & 15;
    int n = ntile * 16 + l15;
    v8f acc = gemm_tile_bf16(sA, h2T, n, lane);

#pragma unroll
    for (int r = 0; r < 8; ++r) {
        int o = mtile * 16 + 8 * half + r;
        float v = acc[r];
        if (o < 36) {
            float inv = rsqrtf(av[o] + EPS_BN);
            v = (v - am[o]) * (inv * ag[o]) + ab[o];
            v = fminf(fmaxf(v, 0.0f), 6.0f);
            int a = o >> 2, comp = o & 3;
            float anc = anchors[a * 4 + comp];
            float bv = (comp < 2) ? (v + anc) : (expf(v) * anc);
            boxes[(a * HW + n) * 4 + comp] = bv;
        } else if (o >= 48 && o < 57) {
            int oi = o - 48;
            float inv = rsqrtf(ov[oi] + EPS_BN);
            v = (v - om[oi]) * (inv * og[oi]) + ob[oi];
            scores[oi * HW + n] = 1.0f / (1.0f + expf(-v));
        }
    }
}

// ---------------------------------------------------------------------------
// Single-block bitonic sort of 16384 keys: key = score_bits<<32 | (~idx)
// (descending score, ascending index on ties; pads sort last).
// ---------------------------------------------------------------------------
__global__ void k_sort(const float* __restrict__ scores,
                       const float* __restrict__ boxes,
                       unsigned long long* __restrict__ keys,
                       unsigned int* __restrict__ sidx,
                       float* __restrict__ bsort)
{
    int tid = threadIdx.x;
    for (int i = tid; i < NSORT; i += 1024) {
        unsigned long long kk = 0ull;
        if (i < NBOX)
            kk = ((unsigned long long)__float_as_uint(scores[i]) << 32)
               | (unsigned long long)(0xFFFFFFFFu - (unsigned)i);
        keys[i] = kk;
    }
    __syncthreads();
    for (int k = 2; k <= NSORT; k <<= 1) {
        for (int j = k >> 1; j > 0; j >>= 1) {
            for (int i = tid; i < NSORT; i += 1024) {
                int ixj = i ^ j;
                if (ixj > i) {
                    unsigned long long a = keys[i], c = keys[ixj];
                    bool desc = ((i & k) == 0);
                    if (desc ? (a < c) : (a > c)) { keys[i] = c; keys[ixj] = a; }
                }
            }
            __syncthreads();
        }
    }
    for (int s = tid; s < NBOX; s += 1024) {
        unsigned int orig = 0xFFFFFFFFu - (unsigned int)(keys[s] & 0xFFFFFFFFull);
        sidx[s] = orig;
        float4 bb = *(const float4*)(boxes + orig * 4);
        *(float4*)(bsort + s * 4) = bb;
    }
}

// ---------------------------------------------------------------------------
// IoU suppression bitmask in sorted order: mask[r][bj] bit c = (j>r && IoU>thr)
// grid (288,288), 32 threads (one wave) per block.
// ---------------------------------------------------------------------------
__global__ void k_mask(const float* __restrict__ bsort,
                       unsigned int* __restrict__ mask)
{
    __shared__ float sb[32 * 4];
    int tid = threadIdx.x;
    int r  = blockIdx.x * 32 + tid;
    int c0 = blockIdx.y * 32;
    {
        float4 bb = *(const float4*)(bsort + (c0 + tid) * 4);
        *(float4*)(sb + tid * 4) = bb;
    }
    __syncthreads();
    float4 bi = *(const float4*)(bsort + r * 4);
    float areaA = (bi.z - bi.x) * (bi.w - bi.y);
    unsigned int bits = 0u;
#pragma unroll 4
    for (int c = 0; c < 32; ++c) {
        int j = c0 + c;
        float bx1 = sb[c * 4 + 0], by1 = sb[c * 4 + 1];
        float bx2 = sb[c * 4 + 2], by2 = sb[c * 4 + 3];
        float areaB = (bx2 - bx1) * (by2 - by1);
        float iw = fmaxf(fminf(bi.z, bx2) - fmaxf(bi.x, bx1), 0.0f);
        float ih = fmaxf(fminf(bi.w, by2) - fmaxf(bi.y, by1), 0.0f);
        float inter = iw * ih;
        float iou = inter / (areaA + areaB - inter);
        if (j > r && iou > IOU_THR) bits |= (1u << c);
    }
    mask[r * MASKW + blockIdx.y] = bits;
}

// ---------------------------------------------------------------------------
// Sequential greedy scan (single block, 288 threads = 9 waves).
// ---------------------------------------------------------------------------
__global__ void k_scan(const unsigned int* __restrict__ mask,
                       const unsigned int* __restrict__ sidx,
                       unsigned int* __restrict__ keep)
{
    __shared__ unsigned int rem[MASKW];
    __shared__ int cur;
    int tid = threadIdx.x;
    if (tid < MASKW) rem[tid] = 0u;
    __syncthreads();
    for (int i = 0; i < NBOX; ++i) {
        if (tid == 0) cur = ((rem[i >> 5] >> (i & 31)) & 1u) ? 0 : 1;
        __syncthreads();
        if (cur) {
            if (tid < MASKW) rem[tid] |= mask[i * MASKW + tid];
            if (tid == 0) keep[sidx[i]] = 1u;
        }
        __syncthreads();
    }
}

// ---------------------------------------------------------------------------
// Final masked writeout: out[i] = [boxes*m, score*m], 9216 x 5 f32.
// ---------------------------------------------------------------------------
__global__ void k_final(const float* __restrict__ boxes,
                        const float* __restrict__ scores,
                        const unsigned int* __restrict__ keep,
                        float* __restrict__ out)
{
    int i = blockIdx.x * blockDim.x + threadIdx.x;
    if (i >= NBOX) return;
    float m = keep[i] ? 1.0f : 0.0f;
    float4 b = *(const float4*)(boxes + i * 4);
    out[i * 5 + 0] = b.x * m;
    out[i * 5 + 1] = b.y * m;
    out[i * 5 + 2] = b.z * m;
    out[i * 5 + 3] = b.w * m;
    out[i * 5 + 4] = scores[i] * m;
}

// ---------------------------------------------------------------------------
extern "C" void kernel_launch(void* const* d_in, const int* in_sizes, int n_in,
                              void* d_out, int out_size, void* d_ws, size_t ws_size,
                              hipStream_t stream)
{
    const float* x    = (const float*)d_in[0];
    const float* anch = (const float*)d_in[1];
    const float* dww  = (const float*)d_in[2];
    const float* pww  = (const float*)d_in[3];
    const float* g1   = (const float*)d_in[4];
    const float* b1   = (const float*)d_in[5];
    const float* m1   = (const float*)d_in[6];
    const float* v1   = (const float*)d_in[7];
    const float* ancw = (const float*)d_in[8];
    const float* ag   = (const float*)d_in[9];
    const float* ab   = (const float*)d_in[10];
    const float* am   = (const float*)d_in[11];
    const float* av   = (const float*)d_in[12];
    const float* objw = (const float*)d_in[13];
    const float* og   = (const float*)d_in[14];
    const float* ob   = (const float*)d_in[15];
    const float* om   = (const float*)d_in[16];
    const float* ov   = (const float*)d_in[17];

    char* ws = (char*)d_ws;
    unsigned short* h1T       = (unsigned short*)(ws + 0);        // 512 KB
    unsigned short* h2T       = (unsigned short*)(ws + 524288);   // 512 KB
    unsigned short* wpw       = (unsigned short*)(ws + 1048576);  // 128 KB
    unsigned short* whd       = (unsigned short*)(ws + 1179648);  // 32 KB
    float* boxes              = (float*)(ws + 1212416);           // 144 KB
    float* scores             = (float*)(ws + 1359872);           // 36 KB
    unsigned long long* keys  = (unsigned long long*)(ws + 1396736); // 128 KB
    unsigned int* sidx        = (unsigned int*)(ws + 1527808);    // 36 KB
    float* bsort              = (float*)(ws + 1564672);           // 144 KB
    unsigned int* keep        = (unsigned int*)(ws + 1712128);    // 36 KB
    unsigned int* mask        = (unsigned int*)(ws + 1748992);    // 10.1 MB
    float* out = (float*)d_out;
    (void)in_sizes; (void)n_in; (void)out_size; (void)ws_size;

    k_prep  <<<256, 256, 0, stream>>>(pww, ancw, objw, wpw, whd, keep);
    k_dwconv<<<1024, 256, 0, stream>>>(x, dww, h1T);
    k_gemm1 <<<128, 256, 0, stream>>>(wpw, h1T, g1, b1, m1, v1, h2T);
    k_heads <<<32, 256, 0, stream>>>(whd, h2T, anch, ag, ab, am, av,
                                     og, ob, om, ov, boxes, scores);
    k_sort  <<<1, 1024, 0, stream>>>(scores, boxes, keys, sidx, bsort);
    k_mask  <<<dim3(288, 288), 32, 0, stream>>>(bsort, mask);
    k_scan  <<<1, 288, 0, stream>>>(mask, sidx, keep);
    k_final <<<36, 256, 0, stream>>>(boxes, scores, keep, out);
}
